// MultiHeadSelfAttention_4604204941406
// MI455X (gfx1250) — compile-verified, matrix operand order checked by reading, not run on hardware
//
#include <hip/hip_runtime.h>

// ---------------- types / constants ----------------
typedef _Float16 h8   __attribute__((ext_vector_type(8)));
typedef _Float16 v16h __attribute__((ext_vector_type(16)));
typedef float    v8f  __attribute__((ext_vector_type(8)));

static constexpr int L  = 2048;
static constexpr int C  = 256;
static constexpr int H  = 8;
static constexpr int HD = 32;
static constexpr int NB = 4;          // the "N" dim of x (B==1)
static constexpr int M  = NB * L;     // 8192 total rows
static constexpr int K3 = 3 * C;      // 768

#define WMMA(a, b, c) \
  __builtin_amdgcn_wmma_f32_16x16x32_f16(false, (a), false, (b), (short)0, (c), false, false)

// Load one 16-f16 A/B fragment slice for lane: two contiguous 8-halfword runs.
// p must already include (row base + k0 + 8*(lane>>4)).
__device__ __forceinline__ v16h load_frag16(const _Float16* p) {
  h8 lo = *(const h8*)(p);
  h8 hi = *(const h8*)(p + 16);
  v16h r;
#pragma unroll
  for (int i = 0; i < 8; ++i) { r[i] = lo[i]; r[i + 8] = hi[i]; }
  return r;
}

// ---------------- f32 -> f16 convert ----------------
__global__ void cvt_f32_f16(const float* __restrict__ s, _Float16* __restrict__ d, int n) {
  int i = blockIdx.x * blockDim.x + threadIdx.x;
  if (i < n) d[i] = (_Float16)s[i];
}

// ---------------- QKV projection GEMM (32x64 tile per wave) ----------------
// qkv[row][o] = sum_c x16[row][c] * w16[o][c] + bias[o]
// scatter: o = sel*256 + h*32 + d ; q,k -> [n][h][l][32] ; v -> [n][h][32][l] (transposed)
__global__ __launch_bounds__(256) void qkv_gemm(
    const _Float16* __restrict__ x16,   // [M][C]
    const _Float16* __restrict__ w16,   // [K3][C]
    const float*    __restrict__ bias,  // [K3]
    _Float16* __restrict__ q16,
    _Float16* __restrict__ k16,
    _Float16* __restrict__ vT16) {
  const int lane = threadIdx.x & 31;
  const int wave = threadIdx.x >> 5;
  const int t  = blockIdx.x * 8 + wave;          // wave-tile id
  const int NT = K3 / 64;                        // 12 col super-tiles
  const int mt = t / NT, nt = t % NT;            // 32-row x 64-col tile
  const int base = (lane >> 4) * 8;
  const int ln   = lane & 15;

  const _Float16* ap[2];
  const _Float16* bp[4];
#pragma unroll
  for (int i = 0; i < 2; ++i) ap[i] = x16 + (mt * 32 + i * 16 + ln) * C;
#pragma unroll
  for (int j = 0; j < 4; ++j) bp[j] = w16 + (nt * 64 + j * 16 + ln) * C;

  v8f acc[2][4] = {};
#pragma unroll
  for (int k0 = 0; k0 < C; k0 += 32) {
    v16h A[2], B[4];
#pragma unroll
    for (int i = 0; i < 2; ++i) A[i] = load_frag16(ap[i] + k0 + base);
#pragma unroll
    for (int j = 0; j < 4; ++j) B[j] = load_frag16(bp[j] + k0 + base);
#pragma unroll
    for (int i = 0; i < 2; ++i)
#pragma unroll
      for (int j = 0; j < 4; ++j) acc[i][j] = WMMA(A[i], B[j], acc[i][j]);
  }

#pragma unroll
  for (int j = 0; j < 4; ++j) {
    const int o    = nt * 64 + j * 16 + ln;
    const float bv = bias[o];
    const int sel  = o >> 8;
    const int h    = (o & 255) >> 5;
    const int d    = o & 31;
#pragma unroll
    for (int i = 0; i < 2; ++i)
#pragma unroll
      for (int r = 0; r < 8; ++r) {
        int row = mt * 32 + i * 16 + r + ((lane >> 4) << 3);
        int n   = row >> 11, l = row & (L - 1);
        _Float16 val = (_Float16)(acc[i][j][r] + bv);
        if (sel == 0)      q16[(((n * H + h) * L) + l) * HD + d] = val;
        else if (sel == 1) k16[(((n * H + h) * L) + l) * HD + d] = val;
        else               vT16[(((n * H + h) * HD) + d) * L + l] = val;
      }
  }
}

// ---------------- flash attention (one wave / 16 rows / head, 64-key chunks) ------
__global__ __launch_bounds__(32) void attn_flash(
    const _Float16* __restrict__ q16,
    const _Float16* __restrict__ k16,
    const _Float16* __restrict__ vT16,
    _Float16* __restrict__ o16) {            // [M][C]
  __shared__ __align__(16) _Float16 pbuf[16 * 64];   // 2 KB probability tile
  const int lane  = threadIdx.x;
  const int bid   = blockIdx.x;              // n*H*(L/16) + h*(L/16) + ltile
  const int ltile = bid & (L / 16 - 1);
  const int h     = (bid >> 7) & (H - 1);
  const int n     = bid >> 10;
  const int base  = (lane >> 4) * 8;
  const int g     = lane >> 4;
  const int ln    = lane & 15;

  const _Float16* qh = q16 + ((n * H + h) * L) * HD;
  const _Float16* kh = k16 + ((n * H + h) * L) * HD;
  const _Float16* vh = vT16 + ((n * H + h) * HD) * L;

  // A-fragment of q rows (16 x 32), loaded once (hd == one WMMA K step)
  v16h qa = load_frag16(qh + (ltile * 16 + ln) * HD + base);

  v8f acc0 = {}, acc1 = {};
  float rowmax[8], rowsum[8];
#pragma unroll
  for (int r = 0; r < 8; ++r) { rowmax[r] = -__builtin_inff(); rowsum[r] = 0.f; }
  const float scale = 1.0f / (float)(HD * HD);   // 1/1024 (q and k each /hd)

  for (int m0 = 0; m0 < L; m0 += 64) {
    if (m0 + 64 < L) {                           // speculative prefetch of next chunk
      __builtin_prefetch(kh + (m0 + 64 + ln) * HD, 0, 0);
      __builtin_prefetch(vh + ln * L + m0 + 64, 0, 0);
    }
    // ---- scores for 64 keys: four 16x16 tiles ----
    v8f s[4];
#pragma unroll
    for (int tt = 0; tt < 4; ++tt) {
      v16h kb = load_frag16(kh + (m0 + tt * 16 + ln) * HD + base);
      v8f z = {};
      s[tt] = WMMA(qa, kb, z);
    }

    __syncthreads();                             // pbuf reuse fence
#pragma unroll
    for (int r = 0; r < 8; ++r) {
      float a0 = s[0][r] * scale, a1 = s[1][r] * scale;
      float a2 = s[2][r] * scale, a3 = s[3][r] * scale;
      float cmax = fmaxf(fmaxf(a0, a1), fmaxf(a2, a3));
#pragma unroll
      for (int w = 1; w < 16; w <<= 1) cmax = fmaxf(cmax, __shfl_xor(cmax, w, 32));
      float mnew = fmaxf(rowmax[r], cmax);
      float f    = __expf(rowmax[r] - mnew);
      rowmax[r]  = mnew;
      float p0 = __expf(a0 - mnew), p1 = __expf(a1 - mnew);
      float p2 = __expf(a2 - mnew), p3 = __expf(a3 - mnew);
      float ps = (p0 + p1) + (p2 + p3);
#pragma unroll
      for (int w = 1; w < 16; w <<= 1) ps += __shfl_xor(ps, w, 32);
      rowsum[r] = rowsum[r] * f + ps;
      acc0[r] *= f; acc1[r] *= f;
      int row = r + 8 * g;                       // C-fragment row for this lane/VGPR
      pbuf[row * 64 + ln]      = (_Float16)p0;
      pbuf[row * 64 + 16 + ln] = (_Float16)p1;
      pbuf[row * 64 + 32 + ln] = (_Float16)p2;
      pbuf[row * 64 + 48 + ln] = (_Float16)p3;
    }
    __syncthreads();

    // ---- o += P(16x64) x V(64x32), two K-steps x two d-halves ----
#pragma unroll
    for (int ks = 0; ks < 2; ++ks) {
      v16h pa  = load_frag16(pbuf + ln * 64 + ks * 32 + base);
      v16h vb0 = load_frag16(vh + ln * L + m0 + ks * 32 + base);
      v16h vb1 = load_frag16(vh + (16 + ln) * L + m0 + ks * 32 + base);
      acc0 = WMMA(pa, vb0, acc0);
      acc1 = WMMA(pa, vb1, acc1);
    }
  }

#pragma unroll
  for (int r = 0; r < 8; ++r) {
    float inv = 1.0f / rowsum[r];
    int row = ltile * 16 + r + 8 * g;
    int gl  = n * L + row;
    o16[gl * C + h * HD + ln]      = (_Float16)(acc0[r] * inv);
    o16[gl * C + h * HD + 16 + ln] = (_Float16)(acc1[r] * inv);
  }
}

// ---------------- output projection GEMM (32x64 tile per wave) ----------------
__global__ __launch_bounds__(256) void out_gemm(
    const _Float16* __restrict__ o16,   // [M][C]
    const _Float16* __restrict__ w16,   // [C][C]
    const float*    __restrict__ bias,  // [C]
    float* __restrict__ out) {          // [M][C]
  const int lane = threadIdx.x & 31;
  const int wave = threadIdx.x >> 5;
  const int t  = blockIdx.x * 8 + wave;
  const int NT = C / 64;                // 4 col super-tiles
  const int mt = t / NT, nt = t % NT;
  const int base = (lane >> 4) * 8;
  const int ln   = lane & 15;

  const _Float16* ap[2];
  const _Float16* bp[4];
#pragma unroll
  for (int i = 0; i < 2; ++i) ap[i] = o16 + (mt * 32 + i * 16 + ln) * C;
#pragma unroll
  for (int j = 0; j < 4; ++j) bp[j] = w16 + (nt * 64 + j * 16 + ln) * C;

  v8f acc[2][4] = {};
#pragma unroll
  for (int k0 = 0; k0 < C; k0 += 32) {
    v16h A[2], B[4];
#pragma unroll
    for (int i = 0; i < 2; ++i) A[i] = load_frag16(ap[i] + k0 + base);
#pragma unroll
    for (int j = 0; j < 4; ++j) B[j] = load_frag16(bp[j] + k0 + base);
#pragma unroll
    for (int i = 0; i < 2; ++i)
#pragma unroll
      for (int j = 0; j < 4; ++j) acc[i][j] = WMMA(A[i], B[j], acc[i][j]);
  }

#pragma unroll
  for (int j = 0; j < 4; ++j) {
    const int o    = nt * 64 + j * 16 + ln;
    const float bv = bias[o];
#pragma unroll
    for (int i = 0; i < 2; ++i)
#pragma unroll
      for (int r = 0; r < 8; ++r) {
        int row = mt * 32 + i * 16 + r + ((lane >> 4) << 3);
        out[row * C + o] = acc[i][j][r] + bv;
      }
  }
}

// ---------------- launch ----------------
extern "C" void kernel_launch(void* const* d_in, const int* in_sizes, int n_in,
                              void* d_out, int out_size, void* d_ws, size_t ws_size,
                              hipStream_t stream) {
  (void)in_sizes; (void)n_in; (void)out_size; (void)ws_size;
  const float* x     = (const float*)d_in[0];   // [1,4,2048,256]
  const float* qkv_w = (const float*)d_in[1];   // [768,256]
  const float* qkv_b = (const float*)d_in[2];   // [768]
  const float* out_w = (const float*)d_in[3];   // [256,256]
  const float* out_b = (const float*)d_in[4];   // [256]
  float* out = (float*)d_out;

  char* ws = (char*)d_ws;
  constexpr size_t SZ_X   = (size_t)M * C * 2;     // 4 MB
  constexpr size_t SZ_QW  = (size_t)K3 * C * 2;
  constexpr size_t SZ_OW  = (size_t)C * C * 2;
  constexpr size_t SZ_QKV = (size_t)M * C * 2;     // per q/k/vT
  _Float16* x16    = (_Float16*)(ws);
  _Float16* qkvw16 = (_Float16*)(ws + SZ_X);
  _Float16* outw16 = (_Float16*)(ws + SZ_X + SZ_QW);
  _Float16* q16    = (_Float16*)(ws + SZ_X + SZ_QW + SZ_OW);
  _Float16* k16    = (_Float16*)(ws + SZ_X + SZ_QW + SZ_OW + SZ_QKV);
  _Float16* vT16   = (_Float16*)(ws + SZ_X + SZ_QW + SZ_OW + 2 * SZ_QKV);
  _Float16* oA16   = (_Float16*)(ws + SZ_X + SZ_QW + SZ_OW + 3 * SZ_QKV);

  // 1) convert inputs/weights to f16
  cvt_f32_f16<<<(M * C + 255) / 256, 256, 0, stream>>>(x, x16, M * C);
  cvt_f32_f16<<<(K3 * C + 255) / 256, 256, 0, stream>>>(qkv_w, qkvw16, K3 * C);
  cvt_f32_f16<<<(C * C + 255) / 256, 256, 0, stream>>>(out_w, outw16, C * C);

  // 2) QKV projection: (256 m-super)*(12 n-super) = 3072 wave-tiles / 8 per block
  qkv_gemm<<<(256 * 12) / 8, 256, 0, stream>>>(x16, qkvw16, qkv_b, q16, k16, vT16);

  // 3) flash attention: 4 n * 8 h * 128 row-tiles = 4096 single-wave blocks
  attn_flash<<<NB * H * (L / 16), 32, 0, stream>>>(q16, k16, vT16, oA16);

  // 4) output projection: (256)*(4) = 1024 wave-tiles / 8 per block
  out_gemm<<<(256 * 4) / 8, 256, 0, stream>>>(oA16, outw16, out_b, out);
}
